// BiLSTMModel_32461362823682
// MI455X (gfx1250) — compile-verified, matrix operand order checked by reading, not run on hardware
//
#include <hip/hip_runtime.h>
#include <hip/hip_bf16.h>

// ---- types for CDNA5 WMMA -------------------------------------------------
typedef __attribute__((ext_vector_type(16))) __bf16 v16bf;
typedef __attribute__((ext_vector_type(8)))  float  v8f;
typedef __attribute__((ext_vector_type(4)))  __bf16 v4bf;

#define Hc   64     // hidden size
#define Gc   256    // 4*H gate width
#define Tc   2000   // sequence length
#define Bc   256    // batch
#define BTc  16     // batch tile per workgroup (== WMMA M)
#define NTHR 256    // 8 waves (wave32)

static __device__ inline float sigmf(float x) { return 1.0f / (1.0f + __expf(-x)); }
static __device__ inline float tanh_fast(float x) { return 2.0f / (1.0f + __expf(-2.0f * x)) - 1.0f; }

// Load one 32x16 bf16 B-fragment (K-chunk of 32 for output column n) from a
// row-major [256 x 64] f32 weight matrix (PyTorch w[n][k]); done once per
// kernel so per-lane scalar gathers are fine.
static __device__ inline v16bf load_Bfrag(const float* __restrict__ w, int n, int kbase, int lane) {
  v16bf r;
  const int hiL = (lane >= 16) ? 1 : 0;
#pragma unroll
  for (int p = 0; p < 16; ++p) {
    const int g = ((p < 8) ? 0 : 2) + hiL;          // 8-element K group
    const int k = kbase + g * 8 + (p & 7);
    r[p] = (__bf16)w[n * Hc + k];
  }
  return r;
}

static __device__ inline float4 sum4(const float* __restrict__ a,
                                     const float* __restrict__ b, int o) {
  float4 r;
  r.x = a[o + 0] + b[o + 0];
  r.y = a[o + 1] + b[o + 1];
  r.z = a[o + 2] + b[o + 2];
  r.w = a[o + 3] + b[o + 3];
  return r;
}

static __device__ inline v8f wmma_bf16(v16bf a, v16bf b, v8f c) {
  // D(f32 16x16) = A(bf16 16x32) * B(bf16 32x16) + C
  return __builtin_amdgcn_wmma_f32_16x16x32_bf16(false, a, false, b, (short)0, c, false, false);
}

// LDS layout (bytes):
//   [0..2048)      e buffer 0, A-chunks 0,1  (K 0..63 of combined A)
//   [2048..4096)   e buffer 1, A-chunks 0,1
//   [4096..6144)   h buffer,   A-chunks 2,3  (K 64..127 of combined A)
//   [8192..24576)  gate pre-activations [16][256] f32
//   [24576..28672) hf  [16][64] f32   (forward last hidden)
//   [28672..32768) hb  [16][64] f32   (backward first hidden)
// A-chunk layout matches the ISA wave32 16-bit A table: [chunk][lane][16 bf16],
// 32 contiguous bytes per lane -> v16bf load = 2x ds_load_b128.

__global__ __launch_bounds__(NTHR)
void bilstm_fused_kernel(const int* __restrict__ x, const float* __restrict__ emb,
                         const float* __restrict__ w_ih_f, const float* __restrict__ w_hh_f,
                         const float* __restrict__ b_ih_f, const float* __restrict__ b_hh_f,
                         const float* __restrict__ w_ih_b,
                         const float* __restrict__ b_ih_b, const float* __restrict__ b_hh_b,
                         const float* __restrict__ w_fc, const float* __restrict__ b_fc,
                         float* __restrict__ out) {
  __shared__ __align__(1024) unsigned char lds[32768];
  float* const gate = (float*)(lds + 8192);
  float* const hf   = (float*)(lds + 24576);
  float* const hb   = (float*)(lds + 28672);

  const int tid  = threadIdx.x;
  const int lane = tid & 31;
  const int wave = tid >> 5;
  const int b0   = blockIdx.x * BTc;

  // per-wave output columns [n0, n0+32): two 16-wide WMMA N tiles
  const int n0    = wave * 32;
  const int ncol0 = n0 + (lane & 15);
  const int ncol1 = ncol0 + 16;

  // ---- forward weights resident in registers as bf16 B fragments ----------
  v16bf Bw0[4], Bw1[4];
  Bw0[0] = load_Bfrag(w_ih_f, ncol0, 0,  lane);
  Bw0[1] = load_Bfrag(w_ih_f, ncol0, 32, lane);
  Bw0[2] = load_Bfrag(w_hh_f, ncol0, 0,  lane);
  Bw0[3] = load_Bfrag(w_hh_f, ncol0, 32, lane);
  Bw1[0] = load_Bfrag(w_ih_f, ncol1, 0,  lane);
  Bw1[1] = load_Bfrag(w_ih_f, ncol1, 32, lane);
  Bw1[2] = load_Bfrag(w_hh_f, ncol1, 0,  lane);
  Bw1[3] = load_Bfrag(w_hh_f, ncol1, 32, lane);

  // ---- embedding-gather thread mapping: (row gr, float4 group gq) ---------
  const int gr = tid >> 4;                 // 0..15 batch row within tile
  const int gq = tid & 15;                 // 16 x float4 = 64 floats per row
  const int ec = gq >> 3;                  // e chunk 0/1 (K block of 32)
  const int kkE = (gq * 4) & 31;
  const int gE  = kkE >> 3;
  const unsigned eStoreOff =
      (unsigned)(gr + ((gE & 1) ? 16 : 0)) * 32u +
      (unsigned)(((gE >> 1) ? 8 : 0) + (kkE & 7)) * 2u;

  // ---- elementwise-cell thread mapping: (row er, hidden cols ej..ej+3) ----
  const int er = tid >> 4;
  const int ej = (tid & 15) * 4;
  const int kkH = ej & 31;
  const int gH  = kkH >> 3;
  const unsigned hStoreAddr =
      (unsigned)(4 + (ej >> 5)) * 1024u +
      (unsigned)(er + ((gH & 1) ? 16 : 0)) * 32u +
      (unsigned)(((gH >> 1) ? 8 : 0) + (kkH & 7)) * 2u;

  // biases folded into the elementwise phase (per-thread, loaded once)
  const float4 bI = sum4(b_ih_f, b_hh_f, 0   + ej);
  const float4 bF = sum4(b_ih_f, b_hh_f, 64  + ej);
  const float4 bG = sum4(b_ih_f, b_hh_f, 128 + ej);
  const float4 bO = sum4(b_ih_f, b_hh_f, 192 + ej);

  // ---- prologue: e_0 -> buffer 0, h = 0 -----------------------------------
  {
    const int idx = x[(b0 + gr) * Tc + 0];
    float4 v = {0.f, 0.f, 0.f, 0.f};
    if (idx != 0) v = *(const float4*)(emb + (size_t)idx * Hc + gq * 4);
    v4bf p; p[0] = (__bf16)v.x; p[1] = (__bf16)v.y; p[2] = (__bf16)v.z; p[3] = (__bf16)v.w;
    *(v4bf*)(lds + (unsigned)ec * 1024u + eStoreOff) = p;
    v4bf z; z[0] = (__bf16)0.f; z[1] = (__bf16)0.f; z[2] = (__bf16)0.f; z[3] = (__bf16)0.f;
    *(v4bf*)(lds + hStoreAddr) = z;
  }
  __syncthreads();

  float4 cst = {0.f, 0.f, 0.f, 0.f};            // cell state lives in registers
  int idxReg = x[(b0 + gr) * Tc + 1];           // index prefetched for t+1

  // software pipeline: e-chunk A fragments for the *current* step are loaded
  // one step ahead (the data lands in LDS one barrier earlier than h).
  v16bf A0 = *(const v16bf*)(lds + 0u    + (unsigned)lane * 32u);
  v16bf A1 = *(const v16bf*)(lds + 1024u + (unsigned)lane * 32u);

  for (int t = 0; t < Tc; ++t) {
    // issue prefetch: embedding row for t+1 (index from last iter), index t+2
    float4 pre = {0.f, 0.f, 0.f, 0.f};
    const bool havePre = (t + 1 < Tc);
    if (havePre && idxReg != 0)
      pre = *(const float4*)(emb + (size_t)idxReg * Hc + gq * 4);
    int idxNext = 0;
    if (t + 2 < Tc) idxNext = x[(b0 + gr) * Tc + (t + 2)];

    // GEMM: gates[16x256] = [e_t | h][16x128] x [w_ih_f ; w_hh_f]^T
    // (A0/A1 = e_t, already in registers; A2/A3 = h, written last step)
    const v16bf A2 = *(const v16bf*)(lds + 4096u + (unsigned)lane * 32u);
    const v16bf A3 = *(const v16bf*)(lds + 5120u + (unsigned)lane * 32u);
    v8f acc0{};                                  // inline-0 C on first WMMA
    v8f acc1{};
    acc0 = wmma_bf16(A0, Bw0[0], acc0);
    acc1 = wmma_bf16(A0, Bw1[0], acc1);
    acc0 = wmma_bf16(A1, Bw0[1], acc0);
    acc1 = wmma_bf16(A1, Bw1[1], acc1);
    acc0 = wmma_bf16(A2, Bw0[2], acc0);
    acc1 = wmma_bf16(A2, Bw1[2], acc1);
    acc0 = wmma_bf16(A3, Bw0[3], acc0);
    acc1 = wmma_bf16(A3, Bw1[3], acc1);

    // C/D layout: VGPR v, lane -> row = v + 8*(lane/16), col = tile + lane%16
    const int rbase = (lane >> 4) << 3;
#pragma unroll
    for (int v = 0; v < 8; ++v) {
      gate[(rbase + v) * Gc + ncol0] = acc0[v];
      gate[(rbase + v) * Gc + ncol1] = acc1[v];
    }

    // stash prefetched e_{t+1} into the other buffer
    if (havePre) {
      v4bf p; p[0] = (__bf16)pre.x; p[1] = (__bf16)pre.y;
      p[2] = (__bf16)pre.z; p[3] = (__bf16)pre.w;
      *(v4bf*)(lds + (unsigned)(((t + 1) & 1) * 2 + ec) * 1024u + eStoreOff) = p;
    }
    idxReg = idxNext;
    __syncthreads();

    // elementwise LSTM cell: gates at cols j, 64+j, 128+j, 192+j (+ biases)
    const float4 gi = *(const float4*)&gate[er * Gc + 0   + ej];
    const float4 gf = *(const float4*)&gate[er * Gc + 64  + ej];
    const float4 gg = *(const float4*)&gate[er * Gc + 128 + ej];
    const float4 go = *(const float4*)&gate[er * Gc + 192 + ej];
    cst.x = sigmf(gf.x + bF.x) * cst.x + sigmf(gi.x + bI.x) * tanh_fast(gg.x + bG.x);
    cst.y = sigmf(gf.y + bF.y) * cst.y + sigmf(gi.y + bI.y) * tanh_fast(gg.y + bG.y);
    cst.z = sigmf(gf.z + bF.z) * cst.z + sigmf(gi.z + bI.z) * tanh_fast(gg.z + bG.z);
    cst.w = sigmf(gf.w + bF.w) * cst.w + sigmf(gi.w + bI.w) * tanh_fast(gg.w + bG.w);
    const float h0v = sigmf(go.x + bO.x) * tanh_fast(cst.x);
    const float h1v = sigmf(go.y + bO.y) * tanh_fast(cst.y);
    const float h2v = sigmf(go.z + bO.z) * tanh_fast(cst.z);
    const float h3v = sigmf(go.w + bO.w) * tanh_fast(cst.w);
    v4bf hp; hp[0] = (__bf16)h0v; hp[1] = (__bf16)h1v;
    hp[2] = (__bf16)h2v; hp[3] = (__bf16)h3v;
    *(v4bf*)(lds + hStoreAddr) = hp;

    // pipeline: pull next step's e-chunk A fragments now (visible since the
    // producer stores landed before this step's first barrier)
    {
      const unsigned ebn = (unsigned)((t + 1) & 1) * 2048u;
      A0 = *(const v16bf*)(lds + ebn + 0u    + (unsigned)lane * 32u);
      A1 = *(const v16bf*)(lds + ebn + 1024u + (unsigned)lane * 32u);
    }
    if (t == Tc - 1) {
      hf[er * Hc + ej + 0] = h0v; hf[er * Hc + ej + 1] = h1v;
      hf[er * Hc + ej + 2] = h2v; hf[er * Hc + ej + 3] = h3v;
    }
    __syncthreads();
  }

  // ---- backward direction: exactly one cell at t = T-1 with h = c = 0 -----
  {
    const int idx = x[(b0 + gr) * Tc + (Tc - 1)];
    float4 v = {0.f, 0.f, 0.f, 0.f};
    if (idx != 0) v = *(const float4*)(emb + (size_t)idx * Hc + gq * 4);
    v4bf p; p[0] = (__bf16)v.x; p[1] = (__bf16)v.y; p[2] = (__bf16)v.z; p[3] = (__bf16)v.w;
    *(v4bf*)(lds + (unsigned)ec * 1024u + eStoreOff) = p;   // buffer 0
  }
  v16bf Bb0[2], Bb1[2];
  Bb0[0] = load_Bfrag(w_ih_b, ncol0, 0,  lane);
  Bb0[1] = load_Bfrag(w_ih_b, ncol0, 32, lane);
  Bb1[0] = load_Bfrag(w_ih_b, ncol1, 0,  lane);
  Bb1[1] = load_Bfrag(w_ih_b, ncol1, 32, lane);
  const float4 cI = sum4(b_ih_b, b_hh_b, 0   + ej);
  const float4 cG = sum4(b_ih_b, b_hh_b, 128 + ej);
  const float4 cO = sum4(b_ih_b, b_hh_b, 192 + ej);
  __syncthreads();
  {
    const v16bf A0b = *(const v16bf*)(lds + 0u    + (unsigned)lane * 32u);
    const v16bf A1b = *(const v16bf*)(lds + 1024u + (unsigned)lane * 32u);
    v8f a0{}, a1{};
    a0 = wmma_bf16(A0b, Bb0[0], a0);
    a1 = wmma_bf16(A0b, Bb1[0], a1);
    a0 = wmma_bf16(A1b, Bb0[1], a0);
    a1 = wmma_bf16(A1b, Bb1[1], a1);
    const int rbase = (lane >> 4) << 3;
#pragma unroll
    for (int v = 0; v < 8; ++v) {
      gate[(rbase + v) * Gc + ncol0] = a0[v];
      gate[(rbase + v) * Gc + ncol1] = a1[v];
    }
  }
  __syncthreads();
  {
    const float4 gi = *(const float4*)&gate[er * Gc + 0   + ej];
    const float4 gg = *(const float4*)&gate[er * Gc + 128 + ej];
    const float4 go = *(const float4*)&gate[er * Gc + 192 + ej];
    // c = sigmoid(i)*tanh(g) (f*c0 == 0);  h = sigmoid(o)*tanh(c)
    const float c0v = sigmf(gi.x + cI.x) * tanh_fast(gg.x + cG.x);
    const float c1v = sigmf(gi.y + cI.y) * tanh_fast(gg.y + cG.y);
    const float c2v = sigmf(gi.z + cI.z) * tanh_fast(gg.z + cG.z);
    const float c3v = sigmf(gi.w + cI.w) * tanh_fast(gg.w + cG.w);
    hb[er * Hc + ej + 0] = sigmf(go.x + cO.x) * tanh_fast(c0v);
    hb[er * Hc + ej + 1] = sigmf(go.y + cO.y) * tanh_fast(c1v);
    hb[er * Hc + ej + 2] = sigmf(go.z + cO.z) * tanh_fast(c2v);
    hb[er * Hc + ej + 3] = sigmf(go.w + cO.w) * tanh_fast(c3v);
  }
  __syncthreads();

  // ---- FC head: out[b, cls] = b_fc[cls] + [hf|hb] . w_fc[cls, :] ----------
  if (tid < BTc * 12) {
    const int r = tid / 12, cls = tid % 12;
    float s = b_fc[cls];
#pragma unroll 8
    for (int j = 0; j < Hc; ++j) s += hf[r * Hc + j] * w_fc[cls * 128 + j];
#pragma unroll 8
    for (int j = 0; j < Hc; ++j) s += hb[r * Hc + j] * w_fc[cls * 128 + 64 + j];
    out[(b0 + r) * 12 + cls] = s;
  }
}

extern "C" void kernel_launch(void* const* d_in, const int* in_sizes, int n_in,
                              void* d_out, int out_size, void* d_ws, size_t ws_size,
                              hipStream_t stream) {
  (void)in_sizes; (void)n_in; (void)out_size; (void)d_ws; (void)ws_size;
  const int*   x      = (const int*)  d_in[0];
  const float* emb    = (const float*)d_in[1];
  const float* w_ih_f = (const float*)d_in[2];
  const float* w_hh_f = (const float*)d_in[3];
  const float* b_ih_f = (const float*)d_in[4];
  const float* b_hh_f = (const float*)d_in[5];
  const float* w_ih_b = (const float*)d_in[6];
  // d_in[7] = w_hh_b: provably unused (backward scan output is its first step,
  // where h0 == 0 so h0 @ w_hh_b^T == 0).
  const float* b_ih_b = (const float*)d_in[8];
  const float* b_hh_b = (const float*)d_in[9];
  const float* w_fc   = (const float*)d_in[10];
  const float* b_fc   = (const float*)d_in[11];
  float* out = (float*)d_out;

  dim3 grid(Bc / BTc), block(NTHR);
  hipLaunchKernelGGL(bilstm_fused_kernel, grid, block, 0, stream,
                     x, emb, w_ih_f, w_hh_f, b_ih_f, b_hh_f,
                     w_ih_b, b_ih_b, b_hh_b, w_fc, b_fc, out);
}
